// MultiHeadCrossAttention_46737834115811
// MI455X (gfx1250) — matrix operand order repaired and checked by reading, not compiled, not run on hardware
//
#include <hip/hip_runtime.h>
#include <hip/hip_bf16.h>
#include <stdint.h>

#define NH     8
#define HD     32
#define DMODEL 256
#define SEQ    512
#define HWN    4096

typedef __attribute__((ext_vector_type(16))) _Float16 v16h;
typedef __attribute__((ext_vector_type(8)))  _Float16 v8h;
typedef __attribute__((ext_vector_type(8)))  float    v8f;
typedef __attribute__((ext_vector_type(4)))  int      v4i;

#if __has_builtin(__builtin_amdgcn_global_load_async_to_lds_b128)
#define HAVE_ASYNC_LDS 1
#else
#define HAVE_ASYNC_LDS 0
#endif

#if __has_builtin(__builtin_amdgcn_s_wait_asynccnt)
#define WAIT_ASYNC(n) __builtin_amdgcn_s_wait_asynccnt(n)
#else
#define WAIT_ASYNC(n) asm volatile("s_wait_asynccnt " #n ::: "memory")
#endif

typedef __attribute__((address_space(1))) v4i g_v4i;   // global (AS1) int4
typedef __attribute__((address_space(3))) v4i l_v4i;   // LDS    (AS3) int4

// 16-byte global -> LDS copy; async (ASYNCcnt) path on CDNA5.
static __device__ __forceinline__ void cp16(const _Float16* g, _Float16* l) {
#if HAVE_ASYNC_LDS
  __builtin_amdgcn_global_load_async_to_lds_b128(
      (g_v4i*)(uintptr_t)g,
      (l_v4i*)(uint32_t)(uintptr_t)l, 0, 0);
#else
  *(uint4*)l = *(const uint4*)g;
#endif
}

static __device__ __forceinline__ v16h cat16(v8h a, v8h b) {
  return __builtin_shufflevector(a, b, 0,1,2,3,4,5,6,7,8,9,10,11,12,13,14,15);
}
static __device__ __forceinline__ v8f wmma16(v16h a, v16h b, v8f c) {
  return __builtin_amdgcn_wmma_f32_16x16x32_f16(false, a, false, b, (short)0, c, false, false);
}
// A fragment 16x32 (f16): lane m = l&15; halves: k = hi*8 + [0..7] and 16 + hi*8 + [0..7]
static __device__ __forceinline__ v16h frag_a(const _Float16* p, int stride, int ln, int hi) {
  const _Float16* row = p + ln * stride;
  v8h lo = *(const v8h*)(row + hi * 8);
  v8h hs = *(const v8h*)(row + 16 + hi * 8);
  return cat16(lo, hs);
}
// B fragment 32x16 (f16): lane n = l&15; halves: k = hi*8 + [0..7] and 16 + hi*8 + [0..7]
static __device__ __forceinline__ v16h frag_b(const _Float16* p, int stride, int ln, int hi) {
  const _Float16* row = p + ln * stride;
  return cat16(*(const v8h*)(row + hi * 8), *(const v8h*)(row + 16 + hi * 8));
}

// ---------------------------------------------------------------------------
// Projection GEMM: Y[m,n] = sum_k X[m,k] * W[n,k] + bias[n], output f16.
// in_mode 0: X row-major (rows x 256) f32.
// in_mode 1: X = key_value (B,256,4096): X[m,k] = kv[(b*256+k)*4096 + hw], m=b*4096+hw.
// out_mode 0: head-major  out[((b*NH+h)*len + mloc)*32 + nl]     (Q, K)
// out_mode 1: dim-major   out[((b*NH+h)*32 + nl)*len + mloc]     (V)
// ---------------------------------------------------------------------------
__global__ __launch_bounds__(256) void proj_kernel(
    const float* __restrict__ X, const float* __restrict__ W,
    const float* __restrict__ bias, _Float16* __restrict__ out,
    int in_mode, int out_mode, int m_shift, int len)
{
  __shared__ __align__(16) _Float16 As[64 * 40];
  __shared__ __align__(16) _Float16 Bs[64 * 40];
  const int tid = threadIdx.x;
  const int m0 = blockIdx.x * 64;
  const int n0 = blockIdx.y * 64;
  const int w = tid >> 5, l = tid & 31, ln = l & 15, hi = l >> 4;
  const int m_sub = (w & 3) * 16, n_sub = (w >> 2) * 32;
  const int bb = m0 >> 12;          // batch (in_mode 1)
  const int hw0 = m0 & (HWN - 1);
  v8f acc0 = {}, acc1 = {};

  for (int k0 = 0; k0 < DMODEL; k0 += 32) {
    if (in_mode == 0) {
      int row = tid >> 2, ks = (tid & 3) * 8;
      const float* src = X + (size_t)(m0 + row) * DMODEL + k0 + ks;
      v8h h;
      #pragma unroll
      for (int i = 0; i < 8; ++i) h[i] = (_Float16)src[i];
      *(v8h*)&As[row * 40 + ks] = h;
    } else {
      int k = tid >> 3, ms = (tid & 7) * 8;
      const float* src = X + (size_t)(bb * DMODEL + k0 + k) * HWN + hw0 + ms;
      #pragma unroll
      for (int i = 0; i < 8; ++i) As[(ms + i) * 40 + k] = (_Float16)src[i];
    }
    {   // Bs[n][k] = W[n0+n][k0+k]
      int row = tid >> 2, ks = (tid & 3) * 8;
      const float* src = W + (size_t)(n0 + row) * DMODEL + k0 + ks;
      v8h h;
      #pragma unroll
      for (int i = 0; i < 8; ++i) h[i] = (_Float16)src[i];
      *(v8h*)&Bs[row * 40 + ks] = h;
    }
    __syncthreads();
    v16h a  = frag_a(As + m_sub * 40, 40, ln, hi);
    v16h b0 = frag_b(Bs + n_sub * 40, 40, ln, hi);
    v16h b1 = frag_b(Bs + (n_sub + 16) * 40, 40, ln, hi);
    acc0 = wmma16(a, b0, acc0);
    acc1 = wmma16(a, b1, acc1);
    __syncthreads();
  }

  const int n_g0 = n0 + n_sub + ln;
  const int n_g1 = n_g0 + 16;
  const float bias0 = bias[n_g0], bias1 = bias[n_g1];
  #pragma unroll
  for (int r = 0; r < 8; ++r) {
    int m_g  = m0 + m_sub + r + 8 * hi;
    int b    = m_g >> m_shift;
    int mloc = m_g & (len - 1);
    {
      int h_ = n_g0 >> 5, nl = n_g0 & 31;
      size_t idx = (out_mode == 0)
        ? ((size_t)(b * NH + h_) * (size_t)len + mloc) * HD + nl
        : ((size_t)(b * NH + h_) * HD + nl) * (size_t)len + mloc;
      out[idx] = (_Float16)(acc0[r] + bias0);
    }
    {
      int h_ = n_g1 >> 5, nl = n_g1 & 31;
      size_t idx = (out_mode == 0)
        ? ((size_t)(b * NH + h_) * (size_t)len + mloc) * HD + nl
        : ((size_t)(b * NH + h_) * HD + nl) * (size_t)len + mloc;
      out[idx] = (_Float16)(acc1[r] + bias1);
    }
  }
}

// ---------------------------------------------------------------------------
// Flash attention: block = one (b,h) x 64 queries, 4 waves (wave32).
// Q head-major f16, K head-major f16, V dim-major f16; online softmax.
// K/V chunks double-buffered in LDS, filled by async global->LDS copies.
// ---------------------------------------------------------------------------
__global__ __launch_bounds__(128) void attn_kernel(
    const _Float16* __restrict__ Q, const _Float16* __restrict__ K,
    const _Float16* __restrict__ V, _Float16* __restrict__ O)
{
  __shared__ __align__(16) _Float16 Ksh[2][64 * 40];    // [key][dim], padded
  __shared__ __align__(16) _Float16 Vsh[2][32 * 72];    // [dim][key], padded
  __shared__ __align__(16) _Float16 Psh[4 * 16 * 72];   // per-wave P tile
  const int bh = blockIdx.y;
  const int q0 = blockIdx.x * 64;
  const int tid = threadIdx.x;
  const int w = tid >> 5, l = tid & 31, ln = l & 15, hi = l >> 4;

  // Q fragment, pre-scaled by 1/sqrt(head_dim)
  v16h qa = frag_a(Q + (size_t)(bh * SEQ + q0 + w * 16) * HD, HD, ln, hi);
  const _Float16 hscale = (_Float16)0.17677669529663689f;
  #pragma unroll
  for (int i = 0; i < 16; ++i) qa[i] = qa[i] * hscale;

  float m_run[8], l_run[8];
  #pragma unroll
  for (int r = 0; r < 8; ++r) { m_run[r] = -3.0e38f; l_run[r] = 0.0f; }
  v8f o0 = {}, o1 = {};
  _Float16* Pw = &Psh[w * 16 * 72];

  const _Float16* kbh = K + (size_t)bh * HWN * HD;
  const _Float16* vbh = V + (size_t)bh * HD * HWN;

  // Per-thread staging addressing (4 x 16B copies per chunk: 2 K + 2 V).
  const int krow0 = tid >> 2,          kseg0 = (tid & 3) * 8;
  const int krow1 = (tid + 128) >> 2,  kseg1 = kseg0;          // (tid+128)&3 == tid&3
  const int vrow0 = tid >> 3,          vseg0 = (tid & 7) * 8;
  const int vrow1 = (tid + 128) >> 3,  vseg1 = vseg0;

#define STAGE_CHUNK(buf, key0)                                                  \
  do {                                                                          \
    cp16(kbh + (size_t)((key0) + krow0) * HD + kseg0,                           \
         &Ksh[buf][krow0 * 40 + kseg0]);                                        \
    cp16(kbh + (size_t)((key0) + krow1) * HD + kseg1,                           \
         &Ksh[buf][krow1 * 40 + kseg1]);                                        \
    cp16(vbh + (size_t)vrow0 * HWN + (key0) + vseg0,                            \
         &Vsh[buf][vrow0 * 72 + vseg0]);                                        \
    cp16(vbh + (size_t)vrow1 * HWN + (key0) + vseg1,                            \
         &Vsh[buf][vrow1 * 72 + vseg1]);                                        \
  } while (0)

  const int NCH = HWN / 64;
  STAGE_CHUNK(0, 0);

  for (int kc = 0; kc < NCH; ++kc) {
    const int buf = kc & 1;
    const _Float16* Kc = Ksh[buf];
    const _Float16* Vc = Vsh[buf];

    if (kc + 1 < NCH) {
      STAGE_CHUNK(buf ^ 1, (kc + 1) * 64);  // prefetch next chunk (async)
      WAIT_ASYNC(4);                        // current chunk's 4 copies landed
    } else {
      WAIT_ASYNC(0);
    }
    __syncthreads();                        // all waves' copies visible

    v8f z = {};
    v8f sc0 = wmma16(qa, frag_b(Kc,           40, ln, hi), z);
    v8f sc1 = wmma16(qa, frag_b(Kc + 16 * 40, 40, ln, hi), z);
    v8f sc2 = wmma16(qa, frag_b(Kc + 32 * 40, 40, ln, hi), z);
    v8f sc3 = wmma16(qa, frag_b(Kc + 48 * 40, 40, ln, hi), z);

    float alpha[8];
    #pragma unroll
    for (int r = 0; r < 8; ++r) {
      float v = fmaxf(fmaxf(sc0[r], sc1[r]), fmaxf(sc2[r], sc3[r]));
      v = fmaxf(v, __shfl_xor(v, 1, 32));
      v = fmaxf(v, __shfl_xor(v, 2, 32));
      v = fmaxf(v, __shfl_xor(v, 4, 32));
      v = fmaxf(v, __shfl_xor(v, 8, 32));
      float mn = fmaxf(m_run[r], v);
      alpha[r] = __expf(m_run[r] - mn);
      m_run[r] = mn;
    }
    float rs[8];
    #pragma unroll
    for (int r = 0; r < 8; ++r) {
      float p0 = __expf(sc0[r] - m_run[r]);
      float p1 = __expf(sc1[r] - m_run[r]);
      float p2 = __expf(sc2[r] - m_run[r]);
      float p3 = __expf(sc3[r] - m_run[r]);
      rs[r] = p0 + p1 + p2 + p3;
      _Float16* pr = Pw + (r + 8 * hi) * 72 + ln;     // C-layout -> LDS tile
      pr[0]  = (_Float16)p0;
      pr[16] = (_Float16)p1;
      pr[32] = (_Float16)p2;
      pr[48] = (_Float16)p3;
    }
    #pragma unroll
    for (int r = 0; r < 8; ++r) {
      float s = rs[r];
      s += __shfl_xor(s, 1, 32);
      s += __shfl_xor(s, 2, 32);
      s += __shfl_xor(s, 4, 32);
      s += __shfl_xor(s, 8, 32);
      l_run[r] = l_run[r] * alpha[r] + s;
      o0[r] *= alpha[r];
      o1[r] *= alpha[r];
    }
    __syncthreads();                                   // P visible before re-read

    v16h pa0 = frag_a(Pw,      72, ln, hi);            // P[:, 0:32]
    v16h pa1 = frag_a(Pw + 32, 72, ln, hi);            // P[:,32:64]
    o0 = wmma16(pa0, frag_b(Vc,                72, ln, hi), o0);   // dims 0-15,  keys 0-31
    o0 = wmma16(pa1, frag_b(Vc + 32,           72, ln, hi), o0);   // dims 0-15,  keys 32-63
    o1 = wmma16(pa0, frag_b(Vc + 16 * 72,      72, ln, hi), o1);   // dims 16-31, keys 0-31
    o1 = wmma16(pa1, frag_b(Vc + 16 * 72 + 32, 72, ln, hi), o1);   // dims 16-31, keys 32-63
    __syncthreads();      // done reading buf before next iter's async overwrites peer buffer
  }
#undef STAGE_CHUNK

  const int b = bh >> 3, h_ = bh & 7;
  #pragma unroll
  for (int r = 0; r < 8; ++r) {
    float inv = 1.0f / l_run[r];
    int q = q0 + w * 16 + r + 8 * hi;
    _Float16* orow = O + (size_t)(b * SEQ + q) * DMODEL + h_ * HD;
    orow[ln]      = (_Float16)(o0[r] * inv);
    orow[16 + ln] = (_Float16)(o1[r] * inv);
  }
}

// ---------------------------------------------------------------------------
// Output projection: d_out = o_ws(f16, 2048x256) @ Wo^T + bo   (f32 out)
// ---------------------------------------------------------------------------
__global__ __launch_bounds__(256) void out_proj_kernel(
    const _Float16* __restrict__ A, const float* __restrict__ W,
    const float* __restrict__ bias, float* __restrict__ out)
{
  __shared__ __align__(16) _Float16 As[64 * 40];
  __shared__ __align__(16) _Float16 Bs[64 * 40];
  const int tid = threadIdx.x;
  const int m0 = blockIdx.x * 64;
  const int n0 = blockIdx.y * 64;
  const int w = tid >> 5, l = tid & 31, ln = l & 15, hi = l >> 4;
  const int m_sub = (w & 3) * 16, n_sub = (w >> 2) * 32;
  v8f acc0 = {}, acc1 = {};

  for (int k0 = 0; k0 < DMODEL; k0 += 32) {
    int row = tid >> 2, ks = (tid & 3) * 8;
    *(v8h*)&As[row * 40 + ks] =
        *(const v8h*)(A + (size_t)(m0 + row) * DMODEL + k0 + ks);
    {
      const float* src = W + (size_t)(n0 + row) * DMODEL + k0 + ks;
      v8h h;
      #pragma unroll
      for (int i = 0; i < 8; ++i) h[i] = (_Float16)src[i];
      *(v8h*)&Bs[row * 40 + ks] = h;
    }
    __syncthreads();
    v16h a  = frag_a(As + m_sub * 40, 40, ln, hi);
    v16h b0 = frag_b(Bs + n_sub * 40, 40, ln, hi);
    v16h b1 = frag_b(Bs + (n_sub + 16) * 40, 40, ln, hi);
    acc0 = wmma16(a, b0, acc0);
    acc1 = wmma16(a, b1, acc1);
    __syncthreads();
  }

  const int n_g0 = n0 + n_sub + ln;
  const int n_g1 = n_g0 + 16;
  const float bias0 = bias[n_g0], bias1 = bias[n_g1];
  #pragma unroll
  for (int r = 0; r < 8; ++r) {
    int m_g = m0 + m_sub + r + 8 * hi;
    out[(size_t)m_g * DMODEL + n_g0] = acc0[r] + bias0;
    out[(size_t)m_g * DMODEL + n_g1] = acc1[r] + bias1;
  }
}

extern "C" void kernel_launch(void* const* d_in, const int* in_sizes, int n_in,
                              void* d_out, int out_size, void* d_ws, size_t ws_size,
                              hipStream_t stream) {
  const float* query     = (const float*)d_in[0];
  const float* key_value = (const float*)d_in[1];
  const float* Wq = (const float*)d_in[2];
  const float* bq = (const float*)d_in[3];
  const float* Wk = (const float*)d_in[4];
  const float* bk = (const float*)d_in[5];
  const float* Wv = (const float*)d_in[6];
  const float* bv = (const float*)d_in[7];
  const float* Wo = (const float*)d_in[8];
  const float* bo = (const float*)d_in[9];
  float* out = (float*)d_out;

  char* ws = (char*)d_ws;
  _Float16* q_ws = (_Float16*)(ws);                     // 1 MB : [bh][s][hd]
  _Float16* k_ws = (_Float16*)(ws + (1u << 20));        // 8 MB : [bh][key][hd]
  _Float16* v_ws = (_Float16*)(ws + 9u * (1u << 20));   // 8 MB : [bh][hd][key]
  _Float16* o_ws = (_Float16*)(ws + 17u * (1u << 20));  // 1 MB : [b*s][256]

  proj_kernel<<<dim3(2048 / 64, 4), 256, 0, stream>>>(query,     Wq, bq, q_ws, 0, 0,  9, SEQ);
  proj_kernel<<<dim3(16384 / 64, 4), 256, 0, stream>>>(key_value, Wk, bk, k_ws, 1, 0, 12, HWN);
  proj_kernel<<<dim3(16384 / 64, 4), 256, 0, stream>>>(key_value, Wv, bv, v_ws, 1, 1, 12, HWN);
  attn_kernel<<<dim3(SEQ / 64, 32), 128, 0, stream>>>(q_ws, k_ws, v_ws, o_ws);
  out_proj_kernel<<<dim3(2048 / 64, 4), 256, 0, stream>>>(o_ws, Wo, bo, out);
}
